// PointerNetworkActor_78941498900747
// MI455X (gfx1250) — matrix operand order repaired and checked by reading, not compiled, not run on hardware
//
#include <hip/hip_runtime.h>
#include <hip/hip_bf16.h>
#include <math.h>

typedef __attribute__((ext_vector_type(16))) _Float16 v16h;
typedef __attribute__((ext_vector_type(8)))  float    v8f;

#define Bq 4
#define Sq 512
#define Hq 128
#define G4H 512   // 4*H

// ---------------- fast gfx1250 transcendentals (native TRANS ops) -------------

__device__ __forceinline__ float fast_tanh(float x) {
#if __has_builtin(__builtin_amdgcn_tanhf)
  return __builtin_amdgcn_tanhf(x);     // v_tanh_f32 (gfx1250 TRANS op)
#else
  return tanhf(x);
#endif
}

__device__ __forceinline__ float fast_exp(float x) {
#if __has_builtin(__builtin_amdgcn_exp2f)
  return __builtin_amdgcn_exp2f(x * 1.4426950408889634f);  // v_exp_f32
#else
  return __expf(x);
#endif
}

__device__ __forceinline__ float fast_rcp(float x) {
#if __has_builtin(__builtin_amdgcn_rcpf)
  return __builtin_amdgcn_rcpf(x);      // v_rcp_f32
#else
  return 1.f / x;
#endif
}

__device__ __forceinline__ float sigf(float x) {
  return fast_rcp(1.f + fast_exp(-x));
}

// ---------------- fragment loaders (CDNA5 WMMA 16x16x32 f16 layouts) ----------

// A-matrix 16x32 f16, row-major source [rows][stride]; lane L: row = L&15,
// group g = L>>4; element e -> k = kb + g*16 + e
__device__ __forceinline__ v16h load_afrag(const _Float16* base, int row0, int stride,
                                           int kb, int lane) {
  int r = row0 + (lane & 15);
  int g = lane >> 4;
  const _Float16* p = base + r * stride + kb + g * 16;
  v16h a;
#pragma unroll
  for (int e = 0; e < 8; ++e) a[e] = p[e];
#pragma unroll
  for (int e = 8; e < 16; ++e) a[e] = p[e];
  return a;
}

// B-matrix 32x16 f16 where B[k][n] = W[n][k], W row-major [N][128].
// lane L: n = n0 + (L&15), g = L>>4; element e -> k = kb + g*16 + e
// (contiguous 32 bytes per lane -> b128 vector loads)
__device__ __forceinline__ v16h load_bfrag(const _Float16* W, int n0, int kb, int lane) {
  int n = n0 + (lane & 15);
  int g = lane >> 4;
  const _Float16* p = W + n * Hq + kb + g * 16;
  v16h b;
#pragma unroll
  for (int e = 0; e < 16; ++e) b[e] = p[e];
  return b;
}

// ---------------- kernel 0: weight conversion + x@Wih0^T + bias precompute ----

__global__ __launch_bounds__(256) void pna_prep_kernel(
    const float* __restrict__ x, const float* __restrict__ Wih0,
    const float* __restrict__ Whh0, const float* __restrict__ bih0,
    const float* __restrict__ bhh0, const float* __restrict__ Wih1,
    const float* __restrict__ Whh1, const float* __restrict__ bih1,
    const float* __restrict__ bhh1, const float* __restrict__ W1w,
    const float* __restrict__ W2w,
    float* __restrict__ xg, _Float16* __restrict__ whh0h,
    _Float16* __restrict__ wih1h, _Float16* __restrict__ whh1h,
    _Float16* __restrict__ w1h, _Float16* __restrict__ w2h,
    float* __restrict__ b1sum) {
  int idx = blockIdx.x * blockDim.x + threadIdx.x;
  if (idx < G4H * Hq) {  // 65536: recurrent / layer-1 weights -> f16
    whh0h[idx] = (_Float16)Whh0[idx];
    wih1h[idx] = (_Float16)Wih1[idx];
    whh1h[idx] = (_Float16)Whh1[idx];
  }
  if (idx < Hq * Hq) {  // 16384: attention projections -> f16
    w1h[idx] = (_Float16)W1w[idx];
    w2h[idx] = (_Float16)W2w[idx];
  }
  if (idx < G4H) b1sum[idx] = bih1[idx] + bhh1[idx];
  if (idx < Sq * Bq * G4H) {  // xg[t][b][n] = b0[n] + x[b,t,:] . Wih0[n,:]
    int t = idx >> 11;
    int b = (idx >> 9) & 3;
    int n = idx & 511;
    const float* xp = x + (b * Sq + t) * 2;
    xg[idx] = bih0[n] + bhh0[n] + xp[0] * Wih0[n * 2] + xp[1] * Wih0[n * 2 + 1];
  }
}

// ---------------- kernel 1: persistent 2-layer LSTM, 1 WG, 16 waves ----------
// Wave w owns gate-column tiles {16w, 256+16w}; all recurrent weights live in
// registers as WMMA B-fragments for the whole 512-step recurrence.

__global__ __launch_bounds__(512) void pna_lstm_kernel(
    const float* __restrict__ xg, const _Float16* __restrict__ whh0,
    const _Float16* __restrict__ wih1, const _Float16* __restrict__ whh1,
    const float* __restrict__ b1sum, _Float16* __restrict__ hq) {
  __shared__ _Float16 h0buf[16 * Hq];  // rows 4..15 stay zero (WMMA padding)
  __shared__ _Float16 h1buf[16 * Hq];
  __shared__ float g0buf[Bq * G4H];
  __shared__ float g1buf[Bq * G4H];

  const int tid  = threadIdx.x;
  const int lane = tid & 31;
  const int w    = tid >> 5;  // wave 0..15
  const int cb0  = w * 16;
  const int cb1  = 256 + w * 16;

  for (int i = tid; i < 16 * Hq; i += 512) {
    h0buf[i] = (_Float16)0.f;
    h1buf[i] = (_Float16)0.f;
  }
  __syncthreads();

  // persistent weight fragments (24 x v16h = 192 VGPRs)
  v16h B0a[4], B0b[4], Bi1a[4], Bi1b[4], Bh1a[4], Bh1b[4];
#pragma unroll
  for (int q = 0; q < 4; ++q) {
    B0a[q]  = load_bfrag(whh0, cb0, q * 32, lane);
    B0b[q]  = load_bfrag(whh0, cb1, q * 32, lane);
    Bi1a[q] = load_bfrag(wih1, cb0, q * 32, lane);
    Bi1b[q] = load_bfrag(wih1, cb1, q * 32, lane);
    Bh1a[q] = load_bfrag(whh1, cb0, q * 32, lane);
    Bh1b[q] = load_bfrag(whh1, cb1, q * 32, lane);
  }
  v8f cb1a, cb1b;
  {
    float bva = b1sum[cb0 + (lane & 15)];
    float bvb = b1sum[cb1 + (lane & 15)];
#pragma unroll
    for (int v = 0; v < 8; ++v) { cb1a[v] = bva; cb1b[v] = bvb; }
  }

  const int bidx = tid >> 7;  // batch row 0..3 for the elementwise update
  const int j    = tid & 127; // hidden unit
  float c0 = 0.f, c1 = 0.f;

  for (int t = 0; t < Sq; ++t) {
    // ---- layer 0 gate pre-activations: g0 = h0 @ Whh0^T + xg[t] ----
    v8f acc0a, acc0b;
#pragma unroll
    for (int v = 0; v < 8; ++v) { acc0a[v] = 0.f; acc0b[v] = 0.f; }
    if (lane < 16) {
#pragma unroll
      for (int v = 0; v < 4; ++v) {
        acc0a[v] = xg[(t * 4 + v) * G4H + cb0 + lane];
        acc0b[v] = xg[(t * 4 + v) * G4H + cb1 + lane];
      }
    }
#pragma unroll
    for (int q = 0; q < 4; ++q) {
      v16h A = load_afrag(h0buf, 0, Hq, q * 32, lane);
      acc0a = __builtin_amdgcn_wmma_f32_16x16x32_f16(false, A, false, B0a[q],
                                                     (short)0, acc0a, false, false);
      acc0b = __builtin_amdgcn_wmma_f32_16x16x32_f16(false, A, false, B0b[q],
                                                     (short)0, acc0b, false, false);
    }
    if (lane < 16) {
#pragma unroll
      for (int v = 0; v < 4; ++v) {
        g0buf[v * G4H + cb0 + lane] = acc0a[v];
        g0buf[v * G4H + cb1 + lane] = acc0b[v];
      }
    }
    __syncthreads();

    // ---- layer 0 cell/hidden update (PyTorch gate order i,f,g,o) ----
    {
      float gi = g0buf[bidx * G4H + j];
      float gf = g0buf[bidx * G4H + 128 + j];
      float gg = g0buf[bidx * G4H + 256 + j];
      float go = g0buf[bidx * G4H + 384 + j];
      c0 = sigf(gf) * c0 + sigf(gi) * fast_tanh(gg);
      float h0v = sigf(go) * fast_tanh(c0);
      h0buf[bidx * Hq + j] = (_Float16)h0v;
    }
    __syncthreads();

    // ---- layer 1 gates: g1 = h0 @ Wih1^T + h1 @ Whh1^T + bias1 ----
    v8f acc1a = cb1a, acc1b = cb1b;
#pragma unroll
    for (int q = 0; q < 4; ++q) {
      v16h A = load_afrag(h0buf, 0, Hq, q * 32, lane);
      acc1a = __builtin_amdgcn_wmma_f32_16x16x32_f16(false, A, false, Bi1a[q],
                                                     (short)0, acc1a, false, false);
      acc1b = __builtin_amdgcn_wmma_f32_16x16x32_f16(false, A, false, Bi1b[q],
                                                     (short)0, acc1b, false, false);
    }
#pragma unroll
    for (int q = 0; q < 4; ++q) {
      v16h A = load_afrag(h1buf, 0, Hq, q * 32, lane);
      acc1a = __builtin_amdgcn_wmma_f32_16x16x32_f16(false, A, false, Bh1a[q],
                                                     (short)0, acc1a, false, false);
      acc1b = __builtin_amdgcn_wmma_f32_16x16x32_f16(false, A, false, Bh1b[q],
                                                     (short)0, acc1b, false, false);
    }
    if (lane < 16) {
#pragma unroll
      for (int v = 0; v < 4; ++v) {
        g1buf[v * G4H + cb0 + lane] = acc1a[v];
        g1buf[v * G4H + cb1 + lane] = acc1b[v];
      }
    }
    __syncthreads();

    // ---- layer 1 update; publish h1(t) for attention ----
    {
      float gi = g1buf[bidx * G4H + j];
      float gf = g1buf[bidx * G4H + 128 + j];
      float gg = g1buf[bidx * G4H + 256 + j];
      float go = g1buf[bidx * G4H + 384 + j];
      c1 = sigf(gf) * c1 + sigf(gi) * fast_tanh(gg);
      float h1v = sigf(go) * fast_tanh(c1);
      h1buf[bidx * Hq + j] = (_Float16)h1v;
      hq[(bidx * Sq + t) * Hq + j] = (_Float16)h1v;
    }
    __syncthreads();
  }
}

// ---------------- kernel 2: attention projections aq = hW1^T+b1, bk = hW2^T+b2
// grid 256 blocks x 256 thr: blk>>7 selects matrix, blk&127 = M-tile, wave = N-tile.

__global__ __launch_bounds__(256) void pna_attproj_kernel(
    const _Float16* __restrict__ hq, const _Float16* __restrict__ w1h,
    const _Float16* __restrict__ w2h, const float* __restrict__ b1,
    const float* __restrict__ b2, float* __restrict__ aq,
    float* __restrict__ bk) {
  const int tid   = threadIdx.x;
  const int lane  = tid & 31;
  const int ntile = tid >> 5;       // 0..7  (N = 128)
  const int blk   = blockIdx.x;
  const int msel  = blk >> 7;
  const int mtile = blk & 127;      // 0..127 (M = 2048)
  const _Float16* W = msel ? w2h : w1h;
  const float* bias = msel ? b2 : b1;
  float* out        = msel ? bk : aq;

  const int n0 = ntile * 16;
  v8f acc;
  float bv = bias[n0 + (lane & 15)];
#pragma unroll
  for (int v = 0; v < 8; ++v) acc[v] = bv;

#pragma unroll
  for (int q = 0; q < 4; ++q) {
    v16h A = load_afrag(hq, mtile * 16, Hq, q * 32, lane);
    v16h B = load_bfrag(W, n0, q * 32, lane);
    acc = __builtin_amdgcn_wmma_f32_16x16x32_f16(false, A, false, B, (short)0,
                                                 acc, false, false);
  }
  const int n = n0 + (lane & 15);
  const int g = lane >> 4;
#pragma unroll
  for (int v = 0; v < 8; ++v)
    out[(mtile * 16 + v + 8 * g) * Hq + n] = acc[v];
}

// ---------------- kernel 3: pairwise tanh scoring + distance + mask + softmax
// one block per (b,i) row; thread = j. 134M native v_tanh_f32 total.

__global__ __launch_bounds__(512) void pna_scores_kernel(
    const float* __restrict__ aq, const float* __restrict__ bk,
    const float* __restrict__ x, const int* __restrict__ np,
    const float* __restrict__ vw, const float* __restrict__ dwp,
    float* __restrict__ out) {
  __shared__ float arow[Hq];
  __shared__ float vsh[Hq];
  __shared__ float red[512];

  const int row = blockIdx.x;      // b*512 + i
  const int b   = row >> 9;
  const int i   = row & 511;
  const int tid = threadIdx.x;     // j

  if (tid < Hq) {
    arow[tid] = aq[row * Hq + tid];
    vsh[tid]  = vw[tid];
  }
  __syncthreads();

  const float* bp = bk + (b * Sq + tid) * Hq;
  float acc = 0.f;
#pragma unroll 4
  for (int h = 0; h < Hq; h += 4) {
    __builtin_prefetch(bp + h + 16, 0, 1);
    float4 bv = *(const float4*)(bp + h);
    acc += vsh[h + 0] * fast_tanh(arow[h + 0] + bv.x);
    acc += vsh[h + 1] * fast_tanh(arow[h + 1] + bv.y);
    acc += vsh[h + 2] * fast_tanh(arow[h + 2] + bv.z);
    acc += vsh[h + 3] * fast_tanh(arow[h + 3] + bv.w);
  }

  float xi0 = x[(b * Sq + i) * 2], xi1 = x[(b * Sq + i) * 2 + 1];
  float xj0 = x[(b * Sq + tid) * 2], xj1 = x[(b * Sq + tid) * 2 + 1];
  float dx = xi0 - xj0, dy = xi1 - xj1;
  float d2 = dx * dx + dy * dy;
  float dist = d2 > 0.f ? sqrtf(d2) : 0.f;
  float score = acc - dwp[0] * dist * 0.01f;
  bool masked = (tid >= np[b]);
  if (masked) score = -__builtin_inff();

  red[tid] = score;
  __syncthreads();
#pragma unroll
  for (int s = 256; s > 0; s >>= 1) {
    if (tid < s) red[tid] = fmaxf(red[tid], red[tid + s]);
    __syncthreads();
  }
  float m = red[0];
  __syncthreads();
  float e = masked ? 0.f : fast_exp(score - m);
  red[tid] = e;
  __syncthreads();
#pragma unroll
  for (int s = 256; s > 0; s >>= 1) {
    if (tid < s) red[tid] += red[tid + s];
    __syncthreads();
  }
  out[row * Sq + tid] = e * fast_rcp(red[0]);
}

// ---------------- host-side launch -------------------------------------------

extern "C" void kernel_launch(void* const* d_in, const int* in_sizes, int n_in,
                              void* d_out, int out_size, void* d_ws, size_t ws_size,
                              hipStream_t stream) {
  const float* x     = (const float*)d_in[0];
  const int*   np    = (const int*)d_in[1];
  const float* Wih0  = (const float*)d_in[2];
  const float* Whh0  = (const float*)d_in[3];
  const float* bih0  = (const float*)d_in[4];
  const float* bhh0  = (const float*)d_in[5];
  const float* Wih1  = (const float*)d_in[6];
  const float* Whh1  = (const float*)d_in[7];
  const float* bih1  = (const float*)d_in[8];
  const float* bhh1  = (const float*)d_in[9];
  const float* W1w   = (const float*)d_in[10];
  const float* W1b   = (const float*)d_in[11];
  const float* W2w   = (const float*)d_in[12];
  const float* W2b   = (const float*)d_in[13];
  const float* vw    = (const float*)d_in[14];
  const float* dw    = (const float*)d_in[15];
  float* out = (float*)d_out;

  char* ws = (char*)d_ws;
  float*     xg    = (float*)(ws + 0);               // 512*4*512*4  = 4,194,304
  _Float16*  hq    = (_Float16*)(ws + 4194304);      // 2048*128*2   =   524,288
  _Float16*  whh0h = (_Float16*)(ws + 4718592);      // 65536*2      =   131,072
  _Float16*  wih1h = (_Float16*)(ws + 4849664);
  _Float16*  whh1h = (_Float16*)(ws + 4980736);
  _Float16*  w1h   = (_Float16*)(ws + 5111808);      // 16384*2      =    32,768
  _Float16*  w2h   = (_Float16*)(ws + 5144576);
  float*     b1sum = (float*)(ws + 5177344);         // 512*4        =     2,048
  float*     aq    = (float*)(ws + 5179392);         // 2048*128*4   = 1,048,576
  float*     bk    = (float*)(ws + 6227968);

  pna_prep_kernel<<<4096, 256, 0, stream>>>(x, Wih0, Whh0, bih0, bhh0, Wih1,
                                            Whh1, bih1, bhh1, W1w, W2w, xg,
                                            whh0h, wih1h, whh1h, w1h, w2h,
                                            b1sum);
  pna_lstm_kernel<<<1, 512, 0, stream>>>(xg, whh0h, wih1h, whh1h, b1sum, hq);
  pna_attproj_kernel<<<256, 256, 0, stream>>>(hq, w1h, w2h, W1b, W2b, aq, bk);
  pna_scores_kernel<<<2048, 512, 0, stream>>>(aq, bk, x, np, vw, dw, out);
}
